// SynthesisNet_66881230733859
// MI455X (gfx1250) — matrix operand order repaired and checked by reading, not compiled, not run on hardware
//
#include <hip/hip_runtime.h>
#include <hip/hip_fp16.h>
#include <stdint.h>

// ---------------------------------------------------------------------------
// StyleGAN2 synthesis net on gfx1250 (MI455X).
// Heavy 512x512x3x3 modulated convs are im2col GEMMs on WMMA
// (v_wmma_f32_16x16x32_f16, f32 accumulate). K dimension is reordered
// tap-major (k = tap*512 + channel) so each 32-K LDS slab has a loop-
// invariant spatial tap offset -> branch-free, mul-free B gather.
// ---------------------------------------------------------------------------

typedef _Float16 h16;
typedef __attribute__((ext_vector_type(16))) _Float16 v16h;
typedef __attribute__((ext_vector_type(8)))  float    v8f;

#define CCH   512
#define BATCH 4
#define KTOT  (CCH * 9)            // 4608 = C*3*3 im2col depth
#define AFFG  0.04419417382415922f // 1/sqrt(512)
#define RGBG  0.04419417382415922f // 1/sqrt(512)
#define SQRT2F 1.4142135623730951f

// ---------------- affine: s = ws @ aw^T * gain + ab  (then * postGain) ------
__global__ void affine_kernel(const float* __restrict__ wsAll,
                              const float* __restrict__ aw,
                              const float* __restrict__ ab,
                              float* __restrict__ out,
                              int wsIdx, float postGain) {
  int o = blockIdx.x * blockDim.x + threadIdx.x;
  int b = blockIdx.y;
  if (o >= CCH) return;
  const float* wv  = wsAll + (size_t)(b * 10 + wsIdx) * CCH;
  const float* awr = aw + (size_t)o * CCH;
  float acc = 0.f;
  for (int i = 0; i < CCH; ++i) acc += wv[i] * awr[i];
  out[b * CCH + o] = (acc * AFFG + ab[o]) * postGain;
}

// ---------------- demod: d[b,o] = rsqrt(sum w^2 s^2 + 1e-8) -----------------
__global__ void demod_kernel(const float* __restrict__ w,
                             const float* __restrict__ styles,
                             float* __restrict__ dcoef) {
  __shared__ float red[256];
  int o = blockIdx.x, b = blockIdx.y, t = threadIdx.x;
  const float* wo = w + (size_t)o * KTOT;
  const float* s  = styles + b * CCH;
  float acc = 0.f;
  for (int k = t; k < KTOT; k += 256) {
    float wv = wo[k];
    float sv = s[k / 9];
    acc += wv * wv * sv * sv;
  }
  red[t] = acc; __syncthreads();
  for (int st = 128; st > 0; st >>= 1) {
    if (t < st) red[t] += red[t + st];
    __syncthreads();
  }
  if (t == 0) dcoef[b * CCH + o] = rsqrtf(red[0] + 1e-8f);
}

// ------- weight cast fp32 -> f16, tap-major K reorder, optional flip --------
// source: w[o][i][ky][kx]  ->  wh[o][tap*512 + i], tap = (f(ky)*3 + f(kx))
__global__ void wcast_kernel(const float* __restrict__ w,
                             h16* __restrict__ wh, int flip) {
  int idx = blockIdx.x * 256 + threadIdx.x;
  if (idx >= CCH * KTOT) return;
  int o = idx / KTOT, rem = idx % KTOT;
  int i = rem / 9, r9 = rem % 9;
  int ky = r9 / 3, kx = r9 % 3;
  int tap = flip ? ((2 - ky) * 3 + (2 - kx)) : r9;
  wh[(size_t)o * KTOT + tap * CCH + i] = (h16)w[idx];
}

// ------- style-modulate input, cast to f16, write zero-padded (W+2)^2 -------
__global__ void modpad_kernel(const float* __restrict__ x,
                              const float* __restrict__ styles,
                              h16* __restrict__ xm, int W) {
  int Wp = W + 2;
  int idx = blockIdx.x * 256 + threadIdx.x;
  if (idx >= BATCH * CCH * Wp * Wp) return;
  int xo = idx % Wp;
  int yo = (idx / Wp) % Wp;
  int ci = (idx / (Wp * Wp)) % CCH;
  int b  = idx / (Wp * Wp * CCH);
  int xi = xo - 1, yi = yo - 1;
  float v = 0.f;
  if (xi >= 0 && xi < W && yi >= 0 && yi < W)
    v = x[((size_t)(b * CCH + ci) * W + yi) * W + xi] * styles[b * CCH + ci];
  xm[idx] = (h16)v;
}

// ------- style-modulate + upfirdn up2 (pad 3/3, 4x4 filter*4) -> f16 --------
// input HxH -> output (2H+2)x(2H+2); following conv uses pad 0, so the
// buffer is already "padded" for all taps.
__global__ void modup_kernel(const float* __restrict__ x,
                             const float* __restrict__ styles,
                             const float* __restrict__ filt,
                             h16* __restrict__ xm, int Hin) {
  int Wu = 2 * Hin + 2;
  int idx = blockIdx.x * 256 + threadIdx.x;
  if (idx >= BATCH * CCH * Wu * Wu) return;
  int xo = idx % Wu;
  int yo = (idx / Wu) % Wu;
  int ci = (idx / (Wu * Wu)) % CCH;
  int b  = idx / (Wu * Wu * CCH);
  const float* xin = x + (size_t)(b * CCH + ci) * Hin * Hin;
  float s = styles[b * CCH + ci];
  float acc = 0.f;
  for (int ty = 0; ty < 4; ++ty) {
    int yy = yo + ty - 3;
    if (yy & 1) continue;
    int iy = yy >> 1;
    if (iy < 0 || iy >= Hin) continue;
    for (int tx = 0; tx < 4; ++tx) {
      int xx = xo + tx - 3;
      if (xx & 1) continue;
      int ix = xx >> 1;
      if (ix < 0 || ix >= Hin) continue;
      acc += filt[ty * 4 + tx] * 4.0f * xin[iy * Hin + ix];
    }
  }
  xm[idx] = (h16)(acc * s);
}

// ---------------- WMMA GEMM conv: out = lrelu(A*B * d + noise + bias) -------
// A = wh [512 x 4608] f16, K tap-major.  B = im2col of padded xm, built as
// 32-K slabs into double-buffered LDS. Wave computes a 32x16 tile (2 acc).
// Block = 8 waves = 256 M rows; grid = (2, N/16, B).
__global__ void __launch_bounds__(256)
gemm_conv_kernel(const h16* __restrict__ wh, const h16* __restrict__ xm,
                 float* __restrict__ xout, const float* __restrict__ dcoef,
                 const float* __restrict__ noise, const float* __restrict__ nsp,
                 const float* __restrict__ bias, int N, int Wout, int Wp) {
  __shared__ unsigned int ldsB[2][256];     // two 16col x 32k f16 slabs, [n][k]

  const int t    = threadIdx.x;
  const int wave = t >> 5;
  const int lane = t & 31;
  const int hl   = lane >> 4;
  const int ln   = lane & 15;
  const int b    = blockIdx.z;
  const int tileM = blockIdx.x * 256 + wave * 32;
  const int tileN = blockIdx.y * 16;

  const unsigned short* xb =
      (const unsigned short*)xm + (size_t)b * CCH * Wp * Wp;
  const h16* aRow0 = wh + (size_t)(tileM + ln) * KTOT;
  const h16* aRow1 = wh + (size_t)(tileM + 16 + ln) * KTOT;

  // B-fill coords: thread t owns LDS word (n = t/16, channel pair 2*(t%16))
  const int bn  = t >> 4;
  const int bkk = (t & 15) << 1;
  const int gnB = tileN + bn;
  const int py  = gnB / Wout;
  const int px  = gnB % Wout;
  const unsigned WpWp = (unsigned)(Wp * Wp);
  const int pbase = py * Wp + px;

  // K is tap-major: k = tap*512 + ci. State below describes the NEXT slab
  // to fill; tap/ciOff are thread-uniform (scalar), toff is per-thread.
  int tap = 0, ciOff = 0;
  int toff = pbase;                          // tap 0 -> (ky,kx) = (0,0)

  // prime slab 0 (tap 0, channels bkk, bkk+1)
  {
    unsigned off = (unsigned)bkk * WpWp + (unsigned)toff;
    ldsB[0][t] = (unsigned)xb[off] | ((unsigned)xb[off + WpWp] << 16);
  }
  ciOff = 32;
  __syncthreads();

  union { v16h v; unsigned int u[8]; } a0, a1, bf;
  v8f acc0 = {}, acc1 = {};
  int cur = 0;
  const int koffB = hl * 8;

  auto loadFrags = [&](int k0) {
    // 16-bit A 16x32 layout: V0-3 K = 2r + 8*hl, V4-7 K = 16 + 2(r-4) + 8*hl
    #pragma unroll
    for (int r = 0; r < 4; ++r) {
      a0.u[r]     = *(const unsigned int*)(aRow0 + k0 + koffB + r * 2);
      a0.u[r + 4] = *(const unsigned int*)(aRow0 + k0 + 16 + koffB + r * 2);
      a1.u[r]     = *(const unsigned int*)(aRow1 + k0 + koffB + r * 2);
      a1.u[r + 4] = *(const unsigned int*)(aRow1 + k0 + 16 + koffB + r * 2);
    }
    // 16-bit B 32x16 layout: V0-3 K = 2r + 8*hl, V4-7 K = 16 + 2(r-4) + 8*hl, col = ln
    #pragma unroll
    for (int r = 0; r < 8; ++r)
      bf.u[r] = ldsB[cur][ln * 16 + (r >> 2) * 8 + hl * 4 + (r & 3)];
  };

  for (int k0 = 0; k0 < KTOT - 32; k0 += 32) {
    loadFrags(k0);

    // fill next slab (k0+32): same tap for the whole slab, channels c, c+1
    {
      unsigned off = (unsigned)(ciOff + bkk) * WpWp + (unsigned)toff;
      ldsB[cur ^ 1][t] = (unsigned)xb[off] | ((unsigned)xb[off + WpWp] << 16);
      ciOff += 32;
      if (ciOff == CCH) {                    // uniform: once per 16 slabs
        ciOff = 0;
        ++tap;
        int ky = (tap > 5) ? 2 : ((tap > 2) ? 1 : 0);
        int kx = tap - ky * 3;
        toff = pbase + ky * Wp + kx;
      }
    }

    acc0 = __builtin_amdgcn_wmma_f32_16x16x32_f16(
        false, a0.v, false, bf.v, (short)0, acc0, false, false);
    acc1 = __builtin_amdgcn_wmma_f32_16x16x32_f16(
        false, a1.v, false, bf.v, (short)0, acc1, false, false);

    __syncthreads();
    cur ^= 1;
  }

  // final slab: no fill, no barrier
  loadFrags(KTOT - 32);
  acc0 = __builtin_amdgcn_wmma_f32_16x16x32_f16(
      false, a0.v, false, bf.v, (short)0, acc0, false, false);
  acc1 = __builtin_amdgcn_wmma_f32_16x16x32_f16(
      false, a1.v, false, bf.v, (short)0, acc1, false, false);

  // fused epilogue: demod, noise, bias, leaky_relu(0.2) * sqrt(2)
  const int   coln = tileN + ln;
  const float nzv  = noise[coln] * nsp[0];
  float* outB = xout + (size_t)b * CCH * N;
  #pragma unroll
  for (int r = 0; r < 8; ++r) {
    int m0 = tileM + hl * 8 + r;             // C/D layout: VGPR r -> M = r + 8*hl
    float v0 = acc0[r] * dcoef[b * CCH + m0] + nzv + bias[m0];
    v0 = (v0 < 0.f ? 0.2f * v0 : v0) * SQRT2F;
    outB[(size_t)m0 * N + coln] = v0;
    int m1 = m0 + 16;
    float v1 = acc1[r] * dcoef[b * CCH + m1] + nzv + bias[m1];
    v1 = (v1 < 0.f ? 0.2f * v1 : v1) * SQRT2F;
    outB[(size_t)m1 * N + coln] = v1;
  }
}

// ---------------- toRGB 1x1 conv (3 outputs), optional add ------------------
__global__ void torgb_kernel(const float* __restrict__ x,
                             const float* __restrict__ styles,
                             const float* __restrict__ w,
                             const float* __restrict__ bias,
                             const float* __restrict__ addBuf, int doAdd,
                             float* __restrict__ out, int N) {
  int n = blockIdx.x * 256 + threadIdx.x;
  int b = blockIdx.y;
  if (n >= N) return;
  const float* xb = x + (size_t)b * CCH * N;
  const float* s  = styles + b * CCH;
  float a0 = 0.f, a1 = 0.f, a2 = 0.f;
  for (int i = 0; i < CCH; ++i) {
    float xv = xb[(size_t)i * N + n] * s[i];
    a0 += w[i] * xv;
    a1 += w[CCH + i] * xv;
    a2 += w[2 * CCH + i] * xv;
  }
  float v0 = a0 + bias[0], v1 = a1 + bias[1], v2 = a2 + bias[2];
  if (doAdd) {
    v0 += addBuf[(size_t)(b * 3 + 0) * N + n];
    v1 += addBuf[(size_t)(b * 3 + 1) * N + n];
    v2 += addBuf[(size_t)(b * 3 + 2) * N + n];
  }
  out[(size_t)(b * 3 + 0) * N + n] = v0;
  out[(size_t)(b * 3 + 1) * N + n] = v1;
  out[(size_t)(b * 3 + 2) * N + n] = v2;
}

// ---------------- image upfirdn up2 (pad 2/2): HxH -> 2Hx2H -----------------
__global__ void imgup_kernel(const float* __restrict__ in,
                             const float* __restrict__ filt,
                             float* __restrict__ out, int Hin) {
  int Ho  = 2 * Hin;
  int idx = blockIdx.x * 256 + threadIdx.x;
  if (idx >= BATCH * 3 * Ho * Ho) return;
  int xo = idx % Ho;
  int yo = (idx / Ho) % Ho;
  int c  = (idx / (Ho * Ho)) % 3;
  int b  = idx / (Ho * Ho * 3);
  const float* xin = in + (size_t)(b * 3 + c) * Hin * Hin;
  float acc = 0.f;
  for (int ty = 0; ty < 4; ++ty) {
    int yy = yo + ty - 2;
    if (yy & 1) continue;
    int iy = yy >> 1;
    if (iy < 0 || iy >= Hin) continue;
    for (int tx = 0; tx < 4; ++tx) {
      int xx = xo + tx - 2;
      if (xx & 1) continue;
      int ix = xx >> 1;
      if (ix < 0 || ix >= Hin) continue;
      acc += filt[ty * 4 + tx] * 4.0f * xin[iy * Hin + ix];
    }
  }
  out[idx] = acc;
}

// ---------------- broadcast learned const to batch --------------------------
__global__ void bcast_kernel(const float* __restrict__ c, float* __restrict__ x) {
  int idx = blockIdx.x * 256 + threadIdx.x;
  if (idx >= BATCH * CCH * 16) return;
  x[idx] = c[idx % (CCH * 16)];
}

// ===========================================================================

struct ConvP { const float *ab, *aw, *b, *ns, *w; };
struct RgbP  { const float *ab, *aw, *b, *w; };

extern "C" void kernel_launch(void* const* d_in, const int* in_sizes, int n_in,
                              void* d_out, int out_size, void* d_ws, size_t ws_size,
                              hipStream_t stream) {
  (void)in_sizes; (void)n_in; (void)out_size; (void)ws_size;

  // ---- parse inputs in JAX pytree (sorted-dict-key) flatten order ----------
  int p = 0;
  auto nxt = [&]() -> const float* { return (const float*)d_in[p++]; };

  const float* filt = nxt();                       // buffers.filter (4x4)
  // buffers.noise: b16_conv0,b16_conv1,b32_conv0,b32_conv1,b4_conv1,
  //                b64_conv0,b64_conv1,b8_conv0,b8_conv1
  const float* nzB16c0 = nxt(); const float* nzB16c1 = nxt();
  const float* nzB32c0 = nxt(); const float* nzB32c1 = nxt();
  const float* nzB4c1  = nxt();
  const float* nzB64c0 = nxt(); const float* nzB64c1 = nxt();
  const float* nzB8c0  = nxt(); const float* nzB8c1  = nxt();

  auto convP = [&]() { ConvP c; c.ab = nxt(); c.aw = nxt(); c.b = nxt();
                       c.ns = nxt(); c.w = nxt(); return c; };
  auto rgbP  = [&]() { RgbP c; c.ab = nxt(); c.aw = nxt(); c.b = nxt();
                       c.w = nxt(); return c; };

  // params blocks in key order: b16, b32, b4, b64, b8, const
  ConvP b16c0 = convP(); ConvP b16c1 = convP(); RgbP b16r = rgbP();
  ConvP b32c0 = convP(); ConvP b32c1 = convP(); RgbP b32r = rgbP();
  ConvP b4c1  = convP();                         RgbP b4r  = rgbP();
  ConvP b64c0 = convP(); ConvP b64c1 = convP(); RgbP b64r = rgbP();
  ConvP b8c0  = convP(); ConvP b8c1  = convP(); RgbP b8r  = rgbP();
  const float* cst = nxt();                      // const (512,4,4)
  const float* wsp = nxt();                      // ws (4,10,512)

  // ---- workspace carve (~90 MB) --------------------------------------------
  char* base = (char*)d_ws; size_t off = 0;
  auto carve = [&](size_t bytes) -> void* {
    void* r = base + off; off += (bytes + 255) & ~(size_t)255; return r;
  };
  float* styles = (float*)carve((size_t)14 * BATCH * CCH * sizeof(float));
  float* dcoef  = (float*)carve((size_t)9  * BATCH * CCH * sizeof(float));
  h16*   wh     = (h16*)  carve((size_t)CCH * KTOT * sizeof(h16));
  h16*   xm     = (h16*)  carve((size_t)BATCH * CCH * 66 * 66 * sizeof(h16));
  float* x0     = (float*)carve((size_t)BATCH * CCH * 64 * 64 * sizeof(float));
  float* x1     = (float*)carve((size_t)BATCH * CCH * 64 * 64 * sizeof(float));
  float* imgA   = (float*)carve((size_t)BATCH * 3 * 64 * 64 * sizeof(float));
  float* imgB   = (float*)carve((size_t)BATCH * 3 * 64 * 64 * sizeof(float));

  // ---- layer schedule ------------------------------------------------------
  ConvP CL[9] = { b4c1, b8c0, b8c1, b16c0, b16c1, b32c0, b32c1, b64c0, b64c1 };
  const float* NZ[9] = { nzB4c1, nzB8c0, nzB8c1, nzB16c0, nzB16c1,
                         nzB32c0, nzB32c1, nzB64c0, nzB64c1 };
  const int WSI[9] = {0,1,2,3,4,5,6,7,8};
  const int RIN[9] = {4,4,8,8,16,16,32,32,64};
  const int UP [9] = {0,1,0,1,0,1,0,1,0};
  RgbP RG[5]  = { b4r, b8r, b16r, b32r, b64r };
  const int RWSI[5] = {1,3,5,7,9};

  // affines: conv layers -> style slots 0..8; torgb -> slots 9..13 (with gain)
  for (int l = 0; l < 9; ++l)
    affine_kernel<<<dim3(2, BATCH), 256, 0, stream>>>(
        wsp, CL[l].aw, CL[l].ab, styles + l * BATCH * CCH, WSI[l], 1.0f);
  for (int r = 0; r < 5; ++r)
    affine_kernel<<<dim3(2, BATCH), 256, 0, stream>>>(
        wsp, RG[r].aw, RG[r].ab, styles + (9 + r) * BATCH * CCH, RWSI[r], RGBG);

  bcast_kernel<<<(BATCH * CCH * 16 + 255) / 256, 256, 0, stream>>>(cst, x0);

  float* xc = x0; float* xn = x1;
  float* imgC = imgA; float* imgN = imgB;
  int rgbIdx = 0;

  for (int l = 0; l < 9; ++l) {
    const float* sl = styles + l * BATCH * CCH;
    float* dl = dcoef + l * BATCH * CCH;

    demod_kernel<<<dim3(CCH, BATCH), 256, 0, stream>>>(CL[l].w, sl, dl);
    wcast_kernel<<<(CCH * KTOT + 255) / 256, 256, 0, stream>>>(CL[l].w, wh, UP[l]);

    int Hin  = RIN[l];
    int Wout = UP[l] ? 2 * Hin     : Hin;
    int Wp   = UP[l] ? 2 * Hin + 2 : Hin + 2;   // padded input width for GEMM
    int totM = BATCH * CCH * Wp * Wp;
    if (UP[l])
      modup_kernel<<<(totM + 255) / 256, 256, 0, stream>>>(xc, sl, filt, xm, Hin);
    else
      modpad_kernel<<<(totM + 255) / 256, 256, 0, stream>>>(xc, sl, xm, Hin);

    int N = Wout * Wout;
    gemm_conv_kernel<<<dim3(2, N / 16, BATCH), 256, 0, stream>>>(
        wh, xm, xn, dl, NZ[l], CL[l].ns, CL[l].b, N, Wout, Wp);
    { float* t = xc; xc = xn; xn = t; }

    if ((l & 1) == 0) {                       // toRGB after layers 0,2,4,6,8
      int res = Wout;
      int Nr  = res * res;
      const float* sr = styles + (9 + rgbIdx) * BATCH * CCH;
      if (l == 0) {
        torgb_kernel<<<dim3((Nr + 255) / 256, BATCH), 256, 0, stream>>>(
            xc, sr, RG[0].w, RG[0].b, imgC, 0, imgC, Nr);
      } else {
        int totU = BATCH * 3 * res * res;
        imgup_kernel<<<(totU + 255) / 256, 256, 0, stream>>>(imgC, filt, imgN, res / 2);
        float* dst = (l == 8) ? (float*)d_out : imgN;
        torgb_kernel<<<dim3((Nr + 255) / 256, BATCH), 256, 0, stream>>>(
            xc, sr, RG[rgbIdx].w, RG[rgbIdx].b, imgN, 1, dst, Nr);
        { float* t = imgC; imgC = imgN; imgN = t; }
      }
      ++rgbIdx;
    }
  }
}
